// PaiConvSmall_51402168599237
// MI455X (gfx1250) — matrix operand order repaired and checked by reading, not compiled, not run on hardware
//
#include <hip/hip_runtime.h>

// PaiConv forward for MI455X (gfx1250), wave32 + WMMA f16 path.
//
// Pipeline per wave (16 points):
//   1) per-lane gather of the exact f-subset needed by its WMMA A-fragment slots
//   2) f32 adjacency mix xn[t,f] = sum_k x[nbr_k, f] * adjw[k,t]   (adjw from v x adjweight)
//   3) ELU -> f16 A fragments (documented dense 16-bit A layout)
//   4) 9 K-chunks x 4 col-tiles of v_wmma_f32_16x16x32_f16 against f16 W (in d_ws)
//   5) +bias, ELU, zero last vertex, store f32 [B,N,64]

typedef __attribute__((ext_vector_type(16))) _Float16 v16h;
typedef __attribute__((ext_vector_type(8)))  float    v8f;
typedef __attribute__((ext_vector_type(4)))  float    v4f;

#define B_ 8
#define N_ 65536
#define K_ 9
#define F_ 32
#define O_ 64
#define D_ (K_ * F_)   // 288

__global__ void w_to_f16_kernel(const float* __restrict__ W, _Float16* __restrict__ Wh) {
    int i = blockIdx.x * 256 + threadIdx.x;
    if (i < O_ * D_) Wh[i] = (_Float16)W[i];
}

__device__ __forceinline__ float elu_f(float x) {
    return x > 0.0f ? x : (__expf(x) - 1.0f);
}

__global__ __launch_bounds__(256)
void paiconv_wmma_kernel(const float* __restrict__ x,          // [B,N,F] f32
                         const int*   __restrict__ nidx,       // [B,N,K] i32
                         const float* __restrict__ v,          // [N,8]   f32
                         const float* __restrict__ adjweight,  // [8,K,K] f32
                         const _Float16* __restrict__ Wh,      // [O,D]   f16 (pre-converted)
                         const float* __restrict__ bias,       // [O]     f32
                         float* __restrict__ out) {            // [B,N,O] f32
    __shared__ float aw[8 * K_ * K_];   // 2592 B broadcast table
    for (int i = threadIdx.x; i < 8 * K_ * K_; i += 256) aw[i] = adjweight[i];
    __syncthreads();

    const int tid  = threadIdx.x;
    const int wave = tid >> 5;
    const int lane = tid & 31;
    const int m    = lane & 15;   // row within 16x16 tiles / output column index
    const int g    = lane >> 4;   // lane half-group

    // 128 points per block (4 waves x ... actually 8 waves x 16 points)
    const int pbase = blockIdx.x * 128 + wave * 16;  // first point of this wave
    const int p     = pbase + m;                     // this lane's point
    const int b     = p >> 16;                       // N == 65536, block stays in one batch
    const int n     = p & (N_ - 1);

    // ---- per-point v row -------------------------------------------------
    float vr[8];
    #pragma unroll
    for (int s = 0; s < 8; ++s) vr[s] = v[(size_t)n * 8 + s];

    // ---- accumulate xn[t, fset(g,j)] over k neighbors --------------------
    // A-fragment element j of chunk c corresponds to feature
    //   f(g,j) = (j<8) ? g*8 + j : 16 + g*8 + (j-8)
    float xnacc[K_][16];
    #pragma unroll
    for (int t = 0; t < K_; ++t)
        #pragma unroll
        for (int j = 0; j < 16; ++j) xnacc[t][j] = 0.0f;

    const int* nrow = nidx + ((size_t)b * N_ + n) * K_;
    for (int k = 0; k < K_; ++k) {
        const int nb = nrow[k];
        const float* xrow = x + ((size_t)b * N_ + nb) * F_;
        // two contiguous 32B runs: f = g*8+0..7 and f = 16+g*8+0..7
        v4f lo0 = *(const v4f*)(xrow + g * 8);
        v4f lo1 = *(const v4f*)(xrow + g * 8 + 4);
        v4f hi0 = *(const v4f*)(xrow + 16 + g * 8);
        v4f hi1 = *(const v4f*)(xrow + 16 + g * 8 + 4);
        float xv[16];
        #pragma unroll
        for (int j = 0; j < 4; ++j) {
            xv[j]      = lo0[j];
            xv[4 + j]  = lo1[j];
            xv[8 + j]  = hi0[j];
            xv[12 + j] = hi1[j];
        }
        // adjw[m][k][t] = sum_s v[n,s] * adjweight[s,k,t]
        float adjwk[K_];
        #pragma unroll
        for (int t = 0; t < K_; ++t) {
            float a = 0.0f;
            #pragma unroll
            for (int s = 0; s < 8; ++s) a = __builtin_fmaf(vr[s], aw[s * 81 + k * 9 + t], a);
            adjwk[t] = a;
        }
        #pragma unroll
        for (int t = 0; t < K_; ++t)
            #pragma unroll
            for (int j = 0; j < 16; ++j)
                xnacc[t][j] = __builtin_fmaf(xv[j], adjwk[t], xnacc[t][j]);
    }

    // ---- ELU + f16 A fragments ------------------------------------------
    v16h afrag[K_];
    #pragma unroll
    for (int c = 0; c < K_; ++c)
        #pragma unroll
        for (int j = 0; j < 16; ++j)
            afrag[c][j] = (_Float16)elu_f(xnacc[c][j]);

    // ---- WMMA: [16 x 288] @ [288 x 64] -> [16 x 64] ----------------------
    // B fragment (dense col-major per ISA): lane holds column o = ct*16 + m,
    // K(g,j) = (j<8) ? g*8 + j : 16 + g*8 + (j-8) within the 32-wide chunk (matches A).
    v8f acc[4] = {v8f{}, v8f{}, v8f{}, v8f{}};
    #pragma unroll
    for (int c = 0; c < K_; ++c) {
        v16h a = afrag[c];
        #pragma unroll
        for (int ct = 0; ct < 4; ++ct) {
            typedef __attribute__((ext_vector_type(8))) _Float16 v8h_;
            const _Float16* wrow = Wh + (size_t)(ct * 16 + m) * D_ + c * F_ + g * 8;
            const v8h_ blo = *(const v8h_*)(wrow);
            const v8h_ bhi = *(const v8h_*)(wrow + 16);
            const v16h bfr = __builtin_shufflevector(blo, bhi, 0, 1, 2, 3, 4, 5, 6, 7, 8, 9, 10, 11, 12, 13, 14, 15);
            acc[ct] = __builtin_amdgcn_wmma_f32_16x16x32_f16(
                false, a, false, bfr, (short)0, acc[ct], false, false);
        }
    }

    // ---- epilogue: +bias, ELU, mask last vertex, store -------------------
    // D layout: VGPR r, lane -> row 8*g + r, col m
    #pragma unroll
    for (int ct = 0; ct < 4; ++ct) {
        const int o = ct * 16 + m;
        const float bo = bias[o];
        #pragma unroll
        for (int r = 0; r < 8; ++r) {
            const int mrow  = g * 8 + r;
            const int nr    = (pbase + mrow) & (N_ - 1);
            float val = elu_f(acc[ct][r] + bo);
            if (nr == N_ - 1) val = 0.0f;   // zero padding vertex (all batches)
            out[((size_t)b * N_ + nr) * O_ + o] = val;
        }
    }
}

extern "C" void kernel_launch(void* const* d_in, const int* in_sizes, int n_in,
                              void* d_out, int out_size, void* d_ws, size_t ws_size,
                              hipStream_t stream) {
    // setup_inputs order: x, t_vertex(unused), neighbor_index, v, adjweight, W, b
    const float* x    = (const float*)d_in[0];
    const int*   nidx = (const int*)  d_in[2];
    const float* v    = (const float*)d_in[3];
    const float* adjw = (const float*)d_in[4];
    const float* W    = (const float*)d_in[5];
    const float* bias = (const float*)d_in[6];
    float* out = (float*)d_out;

    _Float16* Wh = (_Float16*)d_ws;   // 64*288*2 = 36,864 B of scratch

    w_to_f16_kernel<<<(O_ * D_ + 255) / 256, 256, 0, stream>>>(W, Wh);

    const int points = B_ * N_;            // 524288
    const int blocks = points / 128;       // 4096 blocks, 256 threads (8 waves x 16 pts)
    paiconv_wmma_kernel<<<blocks, 256, 0, stream>>>(x, nidx, v, adjw, Wh, bias, out);
}
